// Mamba2Vision_83279415870022
// MI455X (gfx1250) — matrix operand order repaired and checked by reading, not compiled, not run on hardware
//
#include <hip/hip_runtime.h>

// ---------------------------------------------------------------------------
// MI455X (gfx1250) Mamba2-Vision block.
// All heavy math (9 conv1x1 GEMMs + SSD einsums) runs on v_wmma_f32_16x16x32_bf16.
// Activations are kept as bf16 bits (unsigned short) in channel-major layout
// (C, HW) per batch so WMMA B-fragments are contiguous dword loads.
// Accumulation is fp32 in WMMA accumulators. wave32 layouts per CDNA5 ISA.
// GEMM A-tiles are staged via GLOBAL_LOAD_ASYNC_TO_LDS (ASYNCcnt) when the
// toolchain exposes the builtin; synchronous fallback otherwise.
// ---------------------------------------------------------------------------

#if defined(__gfx1250__) && __has_builtin(__builtin_amdgcn_global_load_async_to_lds_b128)
#define USE_ASYNC_LDS 1
#else
#define USE_ASYNC_LDS 0
#endif

typedef __attribute__((ext_vector_type(16))) __bf16 v16bf;
typedef __attribute__((ext_vector_type(8)))  float  v8f;
typedef int vint4 __attribute__((vector_size(16)));                 // V4i pointee type
typedef __attribute__((address_space(1))) vint4 gvint4;             // global V4i
typedef __attribute__((address_space(3))) vint4 svint4;             // LDS V4i

union AFrag { v16bf v; unsigned int u[8]; };

__device__ inline unsigned short f2bf(float f){
  unsigned int u = __builtin_bit_cast(unsigned int, f);
  return (unsigned short)((u + 0x7fffu + ((u >> 16) & 1u)) >> 16);
}
__device__ inline float bf2f(unsigned short s){
  unsigned int u = ((unsigned int)s) << 16;
  return __builtin_bit_cast(float, u);
}

__device__ inline v8f wmma_bf16(v16bf a, v16bf b, v8f c){
  // (neg_a, A, neg_b, B, c_mod, C, reuse_a, reuse_b)
  return __builtin_amdgcn_wmma_f32_16x16x32_bf16(false, a, false, b, (short)0, c, false, false);
}

__device__ inline void wait_async0(){
#if USE_ASYNC_LDS
#if __has_builtin(__builtin_amdgcn_s_wait_asynccnt)
  __builtin_amdgcn_s_wait_asynccnt(0);
#else
  asm volatile("s_wait_asynccnt 0" ::: "memory");
#endif
#endif
}

// A-matrix fragment (16x32 bf16): lane m=lane&15 holds row M=m; half=lane>>4
// selects K = half*16..half*16+15 (contiguous). Source is row-major in K.
__device__ inline v16bf load_frag_a(const unsigned short* base, int ld, int lane){
  AFrag f;
  const int m = lane & 15, half = lane >> 4;
  const unsigned short* row = base + (size_t)m * ld;
  #pragma unroll
  for (int i = 0; i < 8; ++i){
    f.u[i]     = *(const unsigned int*)(row + half*16 + 2*i);
  }
  return f.v;
}
// B-matrix fragment (32x16 bf16): lane holds column N=lane&15; lanes 0-15 carry
// K=0..15, lanes 16-31 carry K=16..31. Source is row-major in K per column
// (i.e. "column n has its K run contiguous"), so 8 contiguous dwords per lane.
__device__ inline v16bf load_frag_b(const unsigned short* base, int ld, int lane){
  AFrag f;
  const int n = lane & 15, half = lane >> 4;
  const unsigned short* row = base + (size_t)n * ld + half*16;
  #pragma unroll
  for (int i = 0; i < 8; ++i) f.u[i] = *(const unsigned int*)(row + 2*i);
  return f.v;
}

// ---------------------------------------------------------------------------
// fp32 -> bf16 bits cast
// ---------------------------------------------------------------------------
__global__ void k_cast(const float* __restrict__ in, unsigned short* __restrict__ out, int n){
  int i = blockIdx.x * 256 + threadIdx.x;
  if (i < n) out[i] = f2bf(in[i]);
}

// ---------------------------------------------------------------------------
// Generic WMMA GEMM:  Out(M,N) = act(A(M,K) @ B(K,N) + bias)
// A: bf16 row-major weights (shared across batch); B: bf16 (K,N) per batch.
// Block tile 64x64, BK=32, 4 waves (128 thr), each wave a 32x32 region (2x2 frags).
// A tile staged by async-to-LDS DMA; B tile transposed through VGPRs.
// ---------------------------------------------------------------------------
__global__ __launch_bounds__(128) void k_gemm(
    const unsigned short* __restrict__ A,
    const unsigned short* __restrict__ B,
    const float* __restrict__ bias,
    void* __restrict__ Out,
    int M, int N, int K,
    long long strideB, long long strideO,
    int act, int outF32)
{
  __shared__ unsigned short sA[64][40];   // [m][k], padded row (80B, 16B aligned)
  __shared__ unsigned short sB[64][40];   // [n][k] (transposed on load)
  const int bx = blockIdx.x, by = blockIdx.y, bz = blockIdx.z;
  const unsigned short* Bp = B + (size_t)bz * strideB;
  const int tid = threadIdx.x, lane = tid & 31, wid = tid >> 5;
  const int wm = wid >> 1, wn = wid & 1;
  const int arow = tid >> 1, aseg = tid & 1;   // A loader role: 64 rows x 2 halves
  const int brow = tid >> 2, bseg = tid & 3;   // B loader role: 32 rows x 4 segs
  v8f acc[2][2] = {};

  for (int k0 = 0; k0 < K; k0 += 32){
    __syncthreads();
    { // A tile 64x32 : straight copy -> async DMA to LDS (ASYNCcnt tracked)
      int gr = by*64 + arow;
      if (gr >= M) gr = by*64;     // clamp: rows >= M only feed store-masked output rows
#if USE_ASYNC_LDS
      unsigned short* src = (unsigned short*)(A + (size_t)gr*K + k0 + aseg*16);
      unsigned short* dst = &sA[arow][aseg*16];
      __builtin_amdgcn_global_load_async_to_lds_b128(
          (gvint4*)(void*)src, (svint4*)(void*)dst, 0, 0);
      __builtin_amdgcn_global_load_async_to_lds_b128(
          (gvint4*)(void*)(src + 8), (svint4*)(void*)(dst + 8), 0, 0);
#else
      const uint4* src = (const uint4*)(A + (size_t)gr*K + k0 + aseg*16);
      uint4 d0 = src[0], d1 = src[1];
      *(uint4*)&sA[arow][aseg*16]     = d0;
      *(uint4*)&sA[arow][aseg*16 + 8] = d1;
#endif
    }
    { // B tile 32x64, stored transposed [n][k]
      const unsigned short* src = Bp + (size_t)(k0 + brow)*N + bx*64 + bseg*16;
      union { uint4 q[2]; unsigned short s[16]; } uu;
      uu.q[0] = *(const uint4*)src;
      uu.q[1] = *(const uint4*)(src + 8);
      #pragma unroll
      for (int j = 0; j < 16; ++j) sB[bseg*16 + j][brow] = uu.s[j];
    }
    if (k0 + 32 < K){ // prefetch next K tile (global_prefetch_b8)
      int gr = by*64 + arow; if (gr >= M) gr = 0;
      __builtin_prefetch((const void*)(A + (size_t)gr*K + k0 + 32 + aseg*16), 0, 1);
      __builtin_prefetch((const void*)(Bp + (size_t)(k0 + 32 + brow)*N + bx*64 + bseg*16), 0, 1);
    }
    wait_async0();                 // s_wait_asynccnt 0 before making LDS visible
    __syncthreads();
    v16bf a0 = load_frag_a(&sA[wm*32][0],      40, lane);
    v16bf a1 = load_frag_a(&sA[wm*32 + 16][0], 40, lane);
    v16bf b0 = load_frag_b(&sB[wn*32][0],      40, lane);
    v16bf b1 = load_frag_b(&sB[wn*32 + 16][0], 40, lane);
    acc[0][0] = wmma_bf16(a0, b0, acc[0][0]);
    acc[0][1] = wmma_bf16(a0, b1, acc[0][1]);
    acc[1][0] = wmma_bf16(a1, b0, acc[1][0]);
    acc[1][1] = wmma_bf16(a1, b1, acc[1][1]);
  }

  const int lm = lane & 15, lh = lane >> 4;
  #pragma unroll
  for (int i = 0; i < 2; ++i)
  #pragma unroll
  for (int j = 0; j < 2; ++j)
  #pragma unroll
  for (int r = 0; r < 8; ++r){
    int row = by*64 + wm*32 + i*16 + r + 8*lh;    // C/D layout: M = r + 8*half
    int col = bx*64 + wn*32 + j*16 + lm;          //             N = lane&15
    if (row < M){
      float v = acc[i][j][r] + (bias ? bias[row] : 0.f);
      if (act == 1) v = v / (1.f + __expf(-v));   // SiLU
      size_t oi = (size_t)bz*strideO + (size_t)row*N + col;
      if (outF32) ((float*)Out)[oi] = v;
      else        ((unsigned short*)Out)[oi] = f2bf(v);
    }
  }
}

// ---------------------------------------------------------------------------
// depthwise 3x3, SAME pad, SiLU.  in/out: (8,512,64,64) bf16
// ---------------------------------------------------------------------------
__global__ void k_dwconv3x3(const unsigned short* __restrict__ in,
                            const float* __restrict__ wdw, const float* __restrict__ bdw,
                            unsigned short* __restrict__ out){
  size_t i = (size_t)blockIdx.x * 256 + threadIdx.x;       // 2^24 elements
  int wp = (int)(i & 63), hp = (int)((i >> 6) & 63);
  int ch = (int)((i >> 12) & 511), b = (int)(i >> 21);
  const unsigned short* p = in + ((size_t)(b*512 + ch)) * 4096;
  const float* wk = wdw + ch*9;
  float acc = bdw[ch];
  #pragma unroll
  for (int kh = 0; kh < 3; ++kh)
  #pragma unroll
  for (int kw = 0; kw < 3; ++kw){
    int hh = hp + kh - 1, ww = wp + kw - 1;
    if (hh >= 0 && hh < 64 && ww >= 0 && ww < 64)
      acc += wk[kh*3 + kw] * bf2f(p[hh*64 + ww]);
  }
  acc = acc / (1.f + __expf(-acc));
  out[i] = f2bf(acc);
}

// ---------------------------------------------------------------------------
// axis attention (softmax over 64-token line, gated context).  axis=0: W, 1: H
// x: (8,256,4096) bf16; qkv: (8,513,4096) bf16; out: (8,256,4096) bf16
// grid (64 lines, 8 batch), 256 threads (one channel each)
// ---------------------------------------------------------------------------
__global__ __launch_bounds__(256) void k_axis_attn(
    const unsigned short* __restrict__ x, const unsigned short* __restrict__ qkv,
    unsigned short* __restrict__ out, int axis)
{
  const int line = blockIdx.x, b = blockIdx.y, tid = threadIdx.x;
  const int stride = axis ? 64 : 1;
  const int base   = axis ? line : line*64;
  __shared__ float sc[64];
  __shared__ float rinv;
  const unsigned short* q = qkv + (size_t)(b*513) * 4096;
  if (tid < 64) sc[tid] = bf2f(q[base + tid*stride]);
  __syncthreads();
  if (tid == 0){
    float mx = sc[0];
    for (int i = 1; i < 64; ++i) mx = fmaxf(mx, sc[i]);
    float s = 0.f;
    for (int i = 0; i < 64; ++i){ float e = __expf(sc[i] - mx); sc[i] = e; s += e; }
    rinv = 1.f / s;
  }
  __syncthreads();
  if (tid < 64) sc[tid] *= rinv;
  __syncthreads();
  const int cch = tid;
  const unsigned short* kch = qkv + (size_t)(b*513 + 1   + cch) * 4096;
  const unsigned short* vch = qkv + (size_t)(b*513 + 257 + cch) * 4096;
  const unsigned short* xch = x   + (size_t)(b*256 + cch) * 4096;
  unsigned short*       och = out + (size_t)(b*256 + cch) * 4096;
  float ctx = 0.f;
  for (int i = 0; i < 64; ++i) ctx += bf2f(kch[base + i*stride]) * sc[i];
  for (int i = 0; i < 64; ++i){
    int a = base + i*stride;
    float vv = bf2f(vch[a]);
    float sg = 1.f / (1.f + __expf(-vv));
    och[a] = f2bf(bf2f(xch[a]) + sg * ctx);
  }
}

// ---------------------------------------------------------------------------
// dt = softplus(xbcdt[512+j] + dt_bias[j]) -> fp32 (8,2,4096)
// ---------------------------------------------------------------------------
__global__ void k_dt(const unsigned short* __restrict__ xbcdt,
                     const float* __restrict__ dt_bias, float* __restrict__ dtb){
  int i = blockIdx.x * 256 + threadIdx.x;                  // 65536
  int t = i & 4095, j = (i >> 12) & 1, b = i >> 13;
  float xv = bf2f(xbcdt[((size_t)(b*514) + 512 + j) * 4096 + t]) + dt_bias[j];
  float sp = (xv > 20.f) ? xv : log1pf(__expf(xv));
  dtb[(size_t)(b*2 + j) * 4096 + t] = sp;
}

// ---------------------------------------------------------------------------
// causal depthwise conv1d (k=4) + SiLU over 512 channels; also emits Xd = xs*dt
// ---------------------------------------------------------------------------
__global__ void k_conv1d(const unsigned short* __restrict__ xbcdt,
                         const float* __restrict__ cw, const float* __restrict__ cb,
                         const float* __restrict__ dtb,
                         unsigned short* __restrict__ xBCc,
                         unsigned short* __restrict__ Xd){
  size_t i = (size_t)blockIdx.x * 256 + threadIdx.x;       // 2^24
  int t = (int)(i & 4095), ch = (int)((i >> 12) & 511), b = (int)(i >> 21);
  const unsigned short* p = xbcdt + (size_t)(b*514 + ch) * 4096;
  float acc = cb[ch];
  #pragma unroll
  for (int j = 0; j < 4; ++j){
    int tt = t - 3 + j;
    if (tt >= 0) acc += cw[ch*4 + j] * bf2f(p[tt]);
  }
  float val = acc / (1.f + __expf(-acc));
  xBCc[(size_t)(b*512 + ch) * 4096 + t] = f2bf(val);
  if (ch < 256){
    int hh = ch >> 7;
    float dtv = dtb[(size_t)(b*2 + hh) * 4096 + t];
    Xd[(size_t)(b*256 + ch) * 4096 + t] = f2bf(val * dtv);
  }
}

// ---------------------------------------------------------------------------
// Transpose B/C groups to token-major (b, g, l, n) for WMMA fragment loads
// ---------------------------------------------------------------------------
__global__ void k_tr_bc(const unsigned short* __restrict__ xBCc,
                        unsigned short* __restrict__ Bt, unsigned short* __restrict__ Ct){
  size_t i = (size_t)blockIdx.x * 256 + threadIdx.x;       // 2^22
  int n = (int)(i & 63), t = (int)((i >> 6) & 4095);
  int g = (int)((i >> 18) & 1), b = (int)(i >> 19);
  Bt[i] = xBCc[((size_t)(b*512) + 256 + g*64 + n) * 4096 + t];
  Ct[i] = xBCc[((size_t)(b*512) + 384 + g*64 + n) * 4096 + t];
}

// ---------------------------------------------------------------------------
// SSD intra-chunk: per (chunk c, head h, batch b) block, 8 waves.
//  - cumsum Acs of dt*A in LDS
//  - st[p,n]  = sum_l exp(Atot-Acs[l]) * Xd[l,p] * B[l,n]        (WMMA 128x64,K=256)
//  - Yd[l,p]  = sum_s ((C[l]·B[s]) * L[l,s]) * Xd[s,p]           (flash-style WMMA)
// Masked score tiles bounce through per-wave LDS to become A-fragments.
// ---------------------------------------------------------------------------
__global__ __launch_bounds__(256) void k_ssd_intra(
    const float* __restrict__ dtb, const float* __restrict__ A_log,
    const unsigned short* __restrict__ Xd, const unsigned short* __restrict__ xBCc,
    const unsigned short* __restrict__ Bt, const unsigned short* __restrict__ Ct,
    float* __restrict__ stBuf, float* __restrict__ AtotBuf,
    float* __restrict__ AcsBuf, float* __restrict__ Y)
{
  const int c = blockIdx.x, h = blockIdx.y, b = blockIdx.z;
  const int tid = threadIdx.x, lane = tid & 31, w = tid >> 5;
  const int m = lane & 15, half = lane >> 4;
  __shared__ float sAcs[256];
  __shared__ unsigned short sS[8][512];   // per-wave 16x32 bf16 bounce tile

  const float Ah = -__expf(A_log[h]);
  sAcs[tid] = dtb[(size_t)(b*2 + h)*4096 + c*256 + tid] * Ah;
  __syncthreads();
  for (int off = 1; off < 256; off <<= 1){            // inclusive scan
    float v = sAcs[tid] + ((tid >= off) ? sAcs[tid - off] : 0.f);
    __syncthreads();
    sAcs[tid] = v;
    __syncthreads();
  }
  const float Atot = sAcs[255];
  if (tid == 0) AtotBuf[(b*16 + c)*2 + h] = Atot;
  AcsBuf[((size_t)(b*16 + c)*2 + h)*256 + tid] = sAcs[tid];

  { // ---- intra-chunk state GEMM: wave w owns p rows [16w, 16w+16)
    const int p0 = w*16;
    const unsigned short* xdBase = Xd + ((size_t)(b*256 + h*128 + p0))*4096 + c*256;
    const unsigned short* bBase  = xBCc + ((size_t)(b*512 + 256 + h*64))*4096 + c*256;
    v8f acc[4] = {};
    for (int kk = 0; kk < 256; kk += 32){
      AFrag fa;   // A = (dec .* Xd)^T rows p, K=l (token-contiguous)
      const unsigned short* rowp = xdBase + (size_t)m*4096 + kk;
      #pragma unroll
      for (int i = 0; i < 8; ++i){
        int kf = half*16 + 2*i;
        unsigned int d = *(const unsigned int*)(rowp + kf);
        float dec0 = __expf(Atot - sAcs[kk + kf]);
        float dec1 = __expf(Atot - sAcs[kk + kf + 1]);
        float x0 = bf2f((unsigned short)(d & 0xffffu)) * dec0;
        float x1 = bf2f((unsigned short)(d >> 16)) * dec1;
        fa.u[i] = ((unsigned int)f2bf(x1) << 16) | f2bf(x0);
      }
      #pragma unroll
      for (int nb = 0; nb < 4; ++nb){
        v16bf fb = load_frag_b(bBase + (size_t)(nb*16)*4096 + kk, 4096, lane);
        acc[nb] = wmma_bf16(fa.v, fb, acc[nb]);
      }
    }
    float* stp = stBuf + ((size_t)((b*16 + c)*2 + h))*128*64;
    #pragma unroll
    for (int nb = 0; nb < 4; ++nb)
    #pragma unroll
    for (int r = 0; r < 8; ++r)
      stp[(size_t)(p0 + r + 8*half)*64 + nb*16 + m] = acc[nb][r];
  }

  { // ---- intra-chunk output Yd: wave w owns l rowblocks {w, w+8} (16 rows each)
    const unsigned short* ctBase = Ct + (((size_t)(b*2 + h))*4096 + c*256)*64;
    const unsigned short* btBase = Bt + (((size_t)(b*2 + h))*4096 + c*256)*64;
    const unsigned short* xdBase = Xd + ((size_t)(b*256 + h*128))*4096 + c*256;
    unsigned short* bounce = &sS[w][0];
    v8f acc[2][8] = {};
    for (int s0 = 0; s0 < 256; s0 += 32){
      v16bf afr[2];
      bool live[2];
      #pragma unroll
      for (int rb = 0; rb < 2; ++rb){
        const int l0 = (w + 8*rb)*16;
        live[rb] = (s0 <= l0 + 15);          // wave-uniform causal skip
        if (!live[rb]) continue;
        #pragma unroll
        for (int sh = 0; sh < 2; ++sh){
          v8f sacc = {};
          #pragma unroll
          for (int kk = 0; kk < 64; kk += 32){
            v16bf fa = load_frag_a(ctBase + (size_t)l0*64 + kk, 64, lane);
            v16bf fb = load_frag_b(btBase + (size_t)(s0 + sh*16)*64 + kk, 64, lane);
            sacc = wmma_bf16(fa, fb, sacc);
          }
          #pragma unroll
          for (int r = 0; r < 8; ++r){        // apply decay mask L[l,s]
            const int l  = l0 + r + 8*half;
            const int sg = s0 + sh*16 + m;
            float v = (sg <= l) ? sacc[r] * __expf(sAcs[l] - sAcs[sg]) : 0.f;
            bounce[(r + 8*half)*32 + sh*16 + m] = f2bf(v);
          }
        }
        afr[rb] = load_frag_a(bounce, 32, lane);   // re-layout as A-fragment
      }
      #pragma unroll
      for (int pb = 0; pb < 8; ++pb){
        v16bf fx = load_frag_b(xdBase + (size_t)(pb*16)*4096 + s0, 4096, lane);
        #pragma unroll
        for (int rb = 0; rb < 2; ++rb)
          if (live[rb]) acc[rb][pb] = wmma_bf16(afr[rb], fx, acc[rb][pb]);
      }
    }
    float* yBase = Y + ((size_t)b*4096 + c*256)*256 + h*128;
    #pragma unroll
    for (int rb = 0; rb < 2; ++rb){
      const int l0 = (w + 8*rb)*16;
      #pragma unroll
      for (int pb = 0; pb < 8; ++pb)
      #pragma unroll
      for (int r = 0; r < 8; ++r)
        yBase[(size_t)(l0 + r + 8*half)*256 + pb*16 + m] = acc[rb][pb][r];
    }
  }
}

// ---------------------------------------------------------------------------
// Inter-chunk state recurrence: Scar[c] = Scar[c-1]*exp(Atot[c-1]) + st[c-1]
// ---------------------------------------------------------------------------
__global__ __launch_bounds__(256) void k_ssd_scan(
    const float* __restrict__ stBuf, const float* __restrict__ AtotBuf,
    float* __restrict__ Scar)
{
  const int h = blockIdx.x, b = blockIdx.y;
  for (int e = threadIdx.x; e < 128*64; e += 256){
    float s = 0.f;
    for (int c = 0; c < 16; ++c){
      size_t idx = ((size_t)((b*16 + c)*2 + h))*128*64 + e;
      Scar[idx] = s;
      s = s * __expf(AtotBuf[(b*16 + c)*2 + h]) + stBuf[idx];
    }
  }
}

// ---------------------------------------------------------------------------
// SSD inter-chunk output: Y += exp(Acs[l]) * (C[l,:] @ Scar[p,:]) + xs*D
// ---------------------------------------------------------------------------
__global__ __launch_bounds__(256) void k_ssd_inter(
    const float* __restrict__ AcsBuf, const unsigned short* __restrict__ Ct,
    const float* __restrict__ Scar, const unsigned short* __restrict__ xBCc,
    const float* __restrict__ Dv, float* __restrict__ Y)
{
  const int c = blockIdx.x, h = blockIdx.y, b = blockIdx.z;
  const int tid = threadIdx.x, lane = tid & 31, w = tid >> 5;
  const int m = lane & 15, half = lane >> 4;
  __shared__ float sAcs[256];
  sAcs[tid] = AcsBuf[((size_t)(b*16 + c)*2 + h)*256 + tid];
  __syncthreads();
  const unsigned short* ctBase = Ct + (((size_t)(b*2 + h))*4096 + c*256)*64;
  const float* scBase = Scar + ((size_t)((b*16 + c)*2 + h))*128*64;
  v8f acc[2][8] = {};
  #pragma unroll
  for (int kk = 0; kk < 64; kk += 32){
    v16bf fa[2];
    #pragma unroll
    for (int rb = 0; rb < 2; ++rb)
      fa[rb] = load_frag_a(ctBase + (size_t)((w + 8*rb)*16)*64 + kk, 64, lane);
    #pragma unroll
    for (int pb = 0; pb < 8; ++pb){
      AFrag fb;  // B = Scar^T : lane owns column p, K=n contiguous fp32
      const float* colp = scBase + (size_t)(pb*16 + m)*64 + kk + half*16;
      #pragma unroll
      for (int i = 0; i < 8; ++i)
        fb.u[i] = ((unsigned int)f2bf(colp[2*i + 1]) << 16) | f2bf(colp[2*i]);
      #pragma unroll
      for (int rb = 0; rb < 2; ++rb)
        acc[rb][pb] = wmma_bf16(fa[rb], fb.v, acc[rb][pb]);
    }
  }
  const float Dh = Dv[h];
  const unsigned short* xsBase = xBCc + ((size_t)(b*512 + h*128))*4096 + c*256;
  float* yBase = Y + ((size_t)b*4096 + c*256)*256 + h*128;
  #pragma unroll
  for (int rb = 0; rb < 2; ++rb){
    const int l0 = (w + 8*rb)*16;
    #pragma unroll
    for (int pb = 0; pb < 8; ++pb)
    #pragma unroll
    for (int r = 0; r < 8; ++r){
      const int l = l0 + r + 8*half;
      const int p = pb*16 + m;
      float yo = __expf(sAcs[l]) * acc[rb][pb][r];
      float xs = bf2f(xsBase[(size_t)p*4096 + l]);
      size_t yi = (size_t)l*256 + p;
      yBase[yi] = yBase[yi] + yo + xs*Dh;
    }
  }
}

// ---------------------------------------------------------------------------
// RMS norm per token; write bf16 into channels [256,512) of concat buffer
// ---------------------------------------------------------------------------
__global__ __launch_bounds__(256) void k_rmsnorm(
    const float* __restrict__ Y, const float* __restrict__ norm_w,
    unsigned short* __restrict__ concat)
{
  const int t = blockIdx.x;               // 0..32767 = b*4096 + tok
  const int ch = threadIdx.x;
  const int b = t >> 12, tok = t & 4095;
  float v = Y[(size_t)t*256 + ch];
  __shared__ float red[256];
  red[ch] = v * v;
  __syncthreads();
  for (int o = 128; o > 0; o >>= 1){
    if (ch < o) red[ch] += red[ch + o];
    __syncthreads();
  }
  float r = rsqrtf(red[0] / 256.f + 1e-5f);
  concat[((size_t)(b*512) + 256 + ch)*4096 + tok] = f2bf(v * r * norm_w[ch]);
}

// ---------------------------------------------------------------------------
// Host launch
// ---------------------------------------------------------------------------
extern "C" void kernel_launch(void* const* d_in, const int* in_sizes, int n_in,
                              void* d_out, int out_size, void* d_ws, size_t ws_size,
                              hipStream_t stream) {
  (void)in_sizes; (void)n_in; (void)out_size; (void)ws_size;
  const float* u         = (const float*)d_in[0];
  const float* sc_w1     = (const float*)d_in[1];
  const float* sc_b1     = (const float*)d_in[2];
  const float* sc_wdw    = (const float*)d_in[3];
  const float* sc_bdw    = (const float*)d_in[4];
  const float* sc_w2     = (const float*)d_in[5];
  const float* sc_b2     = (const float*)d_in[6];
  const float* mc_w      = (const float*)d_in[7];
  const float* mc_b      = (const float*)d_in[8];
  const float* qkvW_w    = (const float*)d_in[9];
  const float* qkvW_b    = (const float*)d_in[10];
  const float* qkvH_w    = (const float*)d_in[11];
  const float* qkvH_b    = (const float*)d_in[12];
  const float* fuseW_w   = (const float*)d_in[13];
  const float* fuseW_b   = (const float*)d_in[14];
  const float* fuseH_w   = (const float*)d_in[15];
  const float* fuseH_b   = (const float*)d_in[16];
  const float* in_proj_w = (const float*)d_in[17];
  const float* conv1d_w  = (const float*)d_in[18];
  const float* conv1d_b  = (const float*)d_in[19];
  const float* dt_bias   = (const float*)d_in[20];
  const float* A_log     = (const float*)d_in[21];
  const float* Dv        = (const float*)d_in[22];
  const float* norm_w    = (const float*)d_in[23];
  const float* outp_w    = (const float*)d_in[24];
  const float* outp_b    = (const float*)d_in[25];

  char* ws = (char*)d_ws;
  size_t off = 0;
  auto alloc = [&](size_t bytes) -> void* {
    void* p = ws + off;
    off += bytes;
    off = (off + 255) & ~(size_t)255;
    return p;
  };

  const size_t NT = 4096;                          // tokens per batch
  unsigned short* u_bf   = (unsigned short*)alloc((size_t)8*512*NT*2);
  unsigned short* w_sc1  = (unsigned short*)alloc((size_t)512*512*2);
  unsigned short* w_sc2  = (unsigned short*)alloc((size_t)256*512*2);
  unsigned short* w_mc   = (unsigned short*)alloc((size_t)256*512*2);
  unsigned short* w_qkvW = (unsigned short*)alloc((size_t)513*256*2);
  unsigned short* w_qkvH = (unsigned short*)alloc((size_t)513*256*2);
  unsigned short* w_fW   = (unsigned short*)alloc((size_t)256*256*2);
  unsigned short* w_fH   = (unsigned short*)alloc((size_t)256*256*2);
  unsigned short* w_inp  = (unsigned short*)alloc((size_t)514*256*2);
  unsigned short* w_out  = (unsigned short*)alloc((size_t)512*512*2);
  unsigned short* z1     = (unsigned short*)alloc((size_t)8*512*NT*2);
  unsigned short* z2     = (unsigned short*)alloc((size_t)8*512*NT*2);
  unsigned short* concat = (unsigned short*)alloc((size_t)8*512*NT*2);
  unsigned short* x1     = (unsigned short*)alloc((size_t)8*256*NT*2);
  unsigned short* bufQ   = (unsigned short*)alloc((size_t)8*513*NT*2);
  unsigned short* attn   = (unsigned short*)alloc((size_t)8*256*NT*2);
  unsigned short* x2     = (unsigned short*)alloc((size_t)8*256*NT*2);
  unsigned short* x3     = (unsigned short*)alloc((size_t)8*256*NT*2);
  unsigned short* xbcdt  = (unsigned short*)alloc((size_t)8*514*NT*2);
  float*          dtb    = (float*)alloc((size_t)8*2*NT*4);
  unsigned short* xBCc   = (unsigned short*)alloc((size_t)8*512*NT*2);
  unsigned short* Xd     = (unsigned short*)alloc((size_t)8*256*NT*2);
  unsigned short* Bt     = (unsigned short*)alloc((size_t)8*2*NT*64*2);
  unsigned short* Ct     = (unsigned short*)alloc((size_t)8*2*NT*64*2);
  float*          stB    = (float*)alloc((size_t)8*16*2*128*64*4);
  float*          Scar   = (float*)alloc((size_t)8*16*2*128*64*4);
  float*          Atot   = (float*)alloc((size_t)8*16*2*4);
  float*          AcsB   = (float*)alloc((size_t)8*16*2*256*4);
  float*          Y      = (float*)alloc((size_t)8*NT*256*4);

  auto cast = [&](const float* src, unsigned short* dst, int n){
    k_cast<<<(n + 255) / 256, 256, 0, stream>>>(src, dst, n);
  };
  auto gemm = [&](const unsigned short* A, const unsigned short* B, const float* bias,
                  void* Out, int M, int K, long long sB, long long sO, int act, int f32){
    dim3 g(4096 / 64, (M + 63) / 64, 8);
    k_gemm<<<g, 128, 0, stream>>>(A, B, bias, Out, M, 4096, K, sB, sO, act, f32);
  };

  // weight / input casts to bf16
  cast(u, u_bf, 8*512*4096);
  cast(sc_w1, w_sc1, 512*512);       cast(sc_w2, w_sc2, 256*512);
  cast(mc_w, w_mc, 256*512);         cast(qkvW_w, w_qkvW, 513*256);
  cast(qkvH_w, w_qkvH, 513*256);     cast(fuseW_w, w_fW, 256*256);
  cast(fuseH_w, w_fH, 256*256);      cast(in_proj_w, w_inp, 514*256);
  cast(outp_w, w_out, 512*512);

  // z branch
  gemm(w_sc1, u_bf, sc_b1, z1, 512, 512, 512LL*4096, 512LL*4096, 1, 0);
  k_dwconv3x3<<<(8*512*4096)/256, 256, 0, stream>>>(z1, sc_wdw, sc_bdw, z2);
  gemm(w_sc2, z2, sc_b2, concat, 256, 512, 512LL*4096, 512LL*4096, 1, 0); // rows 0..255

  // x branch: 1x1 + axis attention (W then H)
  gemm(w_mc, u_bf, mc_b, x1, 256, 512, 512LL*4096, 256LL*4096, 1, 0);
  gemm(w_qkvW, x1, qkvW_b, bufQ, 513, 256, 256LL*4096, 513LL*4096, 0, 0);
  k_axis_attn<<<dim3(64, 8), 256, 0, stream>>>(x1, bufQ, attn, 0);
  gemm(w_fW, attn, fuseW_b, x2, 256, 256, 256LL*4096, 256LL*4096, 0, 0);
  gemm(w_qkvH, x2, qkvH_b, bufQ, 513, 256, 256LL*4096, 513LL*4096, 0, 0);
  k_axis_attn<<<dim3(64, 8), 256, 0, stream>>>(x2, bufQ, attn, 1);
  gemm(w_fH, attn, fuseH_b, x3, 256, 256, 256LL*4096, 256LL*4096, 0, 0);

  // SSD head
  gemm(w_inp, x3, nullptr, xbcdt, 514, 256, 256LL*4096, 514LL*4096, 0, 0);
  k_dt<<<65536/256, 256, 0, stream>>>(xbcdt, dt_bias, dtb);
  k_conv1d<<<(8*512*4096)/256, 256, 0, stream>>>(xbcdt, conv1d_w, conv1d_b, dtb, xBCc, Xd);
  k_tr_bc<<<(8*2*4096*64)/256, 256, 0, stream>>>(xBCc, Bt, Ct);
  k_ssd_intra<<<dim3(16, 2, 8), 256, 0, stream>>>(dtb, A_log, Xd, xBCc, Bt, Ct, stB, Atot, AcsB, Y);
  k_ssd_scan<<<dim3(2, 8), 256, 0, stream>>>(stB, Atot, Scar);
  k_ssd_inter<<<dim3(16, 2, 8), 256, 0, stream>>>(AcsB, Ct, Scar, xBCc, Dv, Y);
  k_rmsnorm<<<32768, 256, 0, stream>>>(Y, norm_w, concat);   // rows 256..511

  // output projection (fp32 result straight to d_out)
  gemm(w_out, concat, outp_b, d_out, 512, 512, 512LL*4096, 512LL*4096, 0, 1);
}
